// HierarchicalOrdering_1460288881208
// MI455X (gfx1250) — compile-verified, matrix operand order checked by reading, not run on hardware
//
#include <hip/hip_runtime.h>
#include <hip/hip_bf16.h>
#include <math.h>

#define BATCH 4
#define NPTS 2048
#define DIM 256
#define KCL 8
#define KM_ITERS 10
#define LAM 0.5f
#define SIG2_INV (1.0f / (100.0f * 100.0f))
#define NEGV (-1e30f)

typedef __attribute__((ext_vector_type(16))) _Float16 v16h;
typedef __attribute__((ext_vector_type(8)))  float    v8f;

union Frag16 { v16h h; unsigned int u[8]; };

// ---------------------------------------------------------------------------
// 1) Row-normalize features to f16 (fn = feat / max(||feat||, 1e-12))
// ---------------------------------------------------------------------------
__global__ void normalize_f16_kernel(const float* __restrict__ feat,
                                     _Float16* __restrict__ fnh) {
    const int row = blockIdx.x;           // B*N rows
    const int d   = threadIdx.x;          // DIM == 256 threads
    const float v = feat[(size_t)row * DIM + d];
    __shared__ float red[DIM];
    red[d] = v * v;
    __syncthreads();
    for (int s = DIM / 2; s > 0; s >>= 1) {
        if (d < s) red[d] += red[d + s];
        __syncthreads();
    }
    const float scale = 1.0f / fmaxf(sqrtf(red[0]), 1e-12f);
    fnh[(size_t)row * DIM + d] = (_Float16)(v * scale);
}

// ---------------------------------------------------------------------------
// 2) k-means: deterministic strided centroid init
// ---------------------------------------------------------------------------
__global__ void kmeans_init_kernel(const float* __restrict__ feat,
                                   float* __restrict__ cent) {
    const int bk = blockIdx.x;            // B*K blocks
    const int b = bk / KCL, k = bk % KCL;
    const int src = k * (NPTS / KCL);
    cent[((size_t)b * KCL + k) * DIM + threadIdx.x] =
        feat[((size_t)b * NPTS + src) * DIM + threadIdx.x];
}

// 3) k-means assignment: ids[i] = argmin_k ||feat_i - cent_k||^2 (first-min)
__global__ void kmeans_assign_kernel(const float* __restrict__ feat,
                                     const float* __restrict__ cent,
                                     int* __restrict__ ids) {
    const int idx = blockIdx.x * blockDim.x + threadIdx.x;  // B*N
    if (idx >= BATCH * NPTS) return;
    const int b = idx / NPTS;
    const float* fr = feat + (size_t)idx * DIM;
    const float* cb = cent + (size_t)b * KCL * DIM;
    float best = 3.4e38f;
    int bk = 0;
    for (int k = 0; k < KCL; ++k) {
        const float* cr = cb + (size_t)k * DIM;
        float s = 0.0f;
        for (int d = 0; d < DIM; ++d) {
            const float df = fr[d] - cr[d];
            s = fmaf(df, df, s);
        }
        if (s < best) { best = s; bk = k; }
    }
    ids[idx] = bk;
}

// 4) k-means update: cent_k = mean of members (unchanged if empty)
__global__ void kmeans_update_kernel(const float* __restrict__ feat,
                                     const int* __restrict__ ids,
                                     float* __restrict__ cent) {
    const int bk = blockIdx.x;            // B*K blocks
    const int b = bk / KCL, k = bk % KCL;
    const int d = threadIdx.x;            // DIM threads
    const float* fb = feat + (size_t)b * NPTS * DIM;
    const int* ib = ids + (size_t)b * NPTS;
    float sum = 0.0f;
    int cnt = 0;
    for (int i = 0; i < NPTS; ++i) {
        if (ib[i] == k) { sum += fb[(size_t)i * DIM + d]; ++cnt; }
    }
    if (cnt > 0)
        cent[((size_t)b * KCL + k) * DIM + d] = sum / (float)cnt;
}

// ---------------------------------------------------------------------------
// 5) Similarity matrix via WMMA: S = fn @ fn^T + LAM * exp(-sqdist/SIG^2)
//    One 16x16 output tile per wave, K=256 -> 8x v_wmma_f32_16x16x32_f16.
//    Per ISA 7.12.2: A lane m holds row m K-pairs; B operand (= fn^T) tile
//    uses the identical per-lane gather with row = output column.
// ---------------------------------------------------------------------------
__device__ __forceinline__ Frag16 load_frag(const _Float16* __restrict__ base,
                                            int row, int k0, int hi16) {
    Frag16 f;
    const _Float16* p = base + (size_t)row * DIM + k0 + hi16 * 8;
#pragma unroll
    for (int v = 0; v < 8; ++v) {
        const int kk = (v < 4) ? (2 * v) : (16 + 2 * (v - 4));
        f.u[v] = *(const unsigned int*)(p + kk);
    }
    return f;
}

__global__ void gram_wmma_kernel(const _Float16* __restrict__ fnh,
                                 const float* __restrict__ coords,
                                 float* __restrict__ S) {
    const int wave = threadIdx.x >> 5;
    const int lane = threadIdx.x & 31;
    const int tileId = blockIdx.x * 8 + wave;       // B*128*128 = 65536 tiles
    const int b  = tileId >> 14;
    const int tm = (tileId >> 7) & 127;
    const int tn = tileId & 127;

    const _Float16* fb = fnh + (size_t)b * NPTS * DIM;
    const int hi16 = lane >> 4;
    const int r    = lane & 15;
    const int rowA = tm * 16 + r;
    const int rowB = tn * 16 + r;

    v8f acc = {};
    for (int k0 = 0; k0 < DIM; k0 += 32) {
        __builtin_prefetch(fb + (size_t)rowA * DIM + k0 + 32, 0, 3);
        __builtin_prefetch(fb + (size_t)rowB * DIM + k0 + 32, 0, 3);
        Frag16 a = load_frag(fb, rowA, k0, hi16);
        Frag16 bm = load_frag(fb, rowB, k0, hi16);
        acc = __builtin_amdgcn_wmma_f32_16x16x32_f16(
            false, a.h, false, bm.h, (short)0, acc, false, false);
    }

    // Fused spatial term + store. C layout: VGPR rr -> m = 16*tm+rr+8*hi16,
    // n = 16*tn + (lane&15).
    const float* cb = coords + (size_t)b * NPTS * 2;
    const int n = tn * 16 + (lane & 15);
    const float cnx = cb[2 * n], cny = cb[2 * n + 1];
    float* Sb = S + (size_t)b * NPTS * NPTS;
#pragma unroll
    for (int rr = 0; rr < 8; ++rr) {
        const int m = tm * 16 + rr + hi16 * 8;
        const float dx = cb[2 * m] - cnx;
        const float dy = cb[2 * m + 1] - cny;
        const float sq = fmaxf(dx * dx + dy * dy, 0.0f);
        Sb[(size_t)m * NPTS + n] = acc[rr] + LAM * __expf(-sq * SIG2_INV);
    }
}

// ---------------------------------------------------------------------------
// 6) conn[i] = sum_j (ids[j]==ids[i]) ? S[i][j] : 0
// ---------------------------------------------------------------------------
__global__ void conn_kernel(const float* __restrict__ S,
                            const int* __restrict__ ids,
                            float* __restrict__ conn) {
    const int row = blockIdx.x;           // B*N rows
    const int b = row / NPTS, i = row % NPTS;
    const int* ib = ids + (size_t)b * NPTS;
    const float* Sr = S + ((size_t)b * NPTS + i) * NPTS;
    const int myid = ib[i];
    float s = 0.0f;
    for (int j = threadIdx.x; j < NPTS; j += 256)
        if (ib[j] == myid) s += Sr[j];
    __shared__ float red[256];
    red[threadIdx.x] = s;
    __syncthreads();
    for (int st = 128; st > 0; st >>= 1) {
        if (threadIdx.x < st) red[threadIdx.x] += red[threadIdx.x + st];
        __syncthreads();
    }
    if (threadIdx.x == 0) conn[row] = red[0];
}

// ---------------------------------------------------------------------------
// 7) Sequential hierarchical ordering: one 1024-thread WG per batch.
//    Each of N steps: cmin over unvisited ids, score = row/conn, first-argmax.
// ---------------------------------------------------------------------------
__global__ void __launch_bounds__(1024)
order_kernel(const float* __restrict__ S,
             const float* __restrict__ conn,
             const int* __restrict__ ids,
             int* __restrict__ order) {
    const int b = blockIdx.x;
    const float* Sb = S + (size_t)b * NPTS * NPTS;
    const float* cb = conn + (size_t)b * NPTS;
    const int* ib = ids + (size_t)b * NPTS;
    int* ob = order + (size_t)b * NPTS;

    __shared__ unsigned char visited[NPTS];
    __shared__ float rv[1024];
    __shared__ int   ri[1024];
    __shared__ int   rmin[1024];
    __shared__ int   s_cur;

    const int tid = threadIdx.x;
    visited[tid] = 0;
    visited[tid + 1024] = 0;
    if (tid == 0) s_cur = -1;
    __syncthreads();

    for (int t = 0; t < NPTS; ++t) {
        const int cur = s_cur;
        // cmin = min over unvisited ids (visited -> K)
        const int m0 = visited[tid] ? KCL : ib[tid];
        const int m1 = visited[tid + 1024] ? KCL : ib[tid + 1024];
        rmin[tid] = min(m0, m1);
        __syncthreads();
        for (int s = 512; s > 0; s >>= 1) {
            if (tid < s) rmin[tid] = min(rmin[tid], rmin[tid + s]);
            __syncthreads();
        }
        const int cmin = rmin[0];
        const bool cont = (cur >= 0) && (ib[cur] == cmin);
        const float* rowp = Sb + (size_t)max(cur, 0) * NPTS;

        float bestV = NEGV;
        int bestI = NPTS;
#pragma unroll
        for (int rep = 0; rep < 2; ++rep) {
            const int i = tid + rep * 1024;
            const bool elig = (!visited[i]) && (ib[i] == cmin);
            const float sc = elig ? (cont ? rowp[i] : cb[i]) : NEGV;
            if (sc > bestV || (sc == bestV && i < bestI)) { bestV = sc; bestI = i; }
        }
        rv[tid] = bestV;
        ri[tid] = bestI;
        __syncthreads();
        for (int s = 512; s > 0; s >>= 1) {
            if (tid < s) {
                const float v2 = rv[tid + s];
                const int i2 = ri[tid + s];
                if (v2 > rv[tid] || (v2 == rv[tid] && i2 < ri[tid])) {
                    rv[tid] = v2;
                    ri[tid] = i2;
                }
            }
            __syncthreads();
        }
        if (tid == 0) {
            const int nxt = ri[0];
            ob[t] = nxt;
            visited[nxt] = 1;
            s_cur = nxt;
        }
        __syncthreads();
    }
}

// ---------------------------------------------------------------------------
// 8) Gather reordered features + emit order indices (as exact floats)
// ---------------------------------------------------------------------------
__global__ void gather_kernel(const float* __restrict__ feat,
                              const int* __restrict__ order,
                              float* __restrict__ out,
                              float* __restrict__ out_idx) {
    const int row = blockIdx.x;           // B*N
    const int b = row / NPTS;
    const int src = order[row];
    const float* fr = feat + ((size_t)b * NPTS + src) * DIM;
    float* orow = out + (size_t)row * DIM;
    for (int d = threadIdx.x; d < DIM; d += 256) orow[d] = fr[d];
    if (threadIdx.x == 0) out_idx[row] = (float)src;
}

// ---------------------------------------------------------------------------
extern "C" void kernel_launch(void* const* d_in, const int* in_sizes, int n_in,
                              void* d_out, int out_size, void* d_ws, size_t ws_size,
                              hipStream_t stream) {
    const float* feat   = (const float*)d_in[0];   // (B, N, D) f32
    const float* coords = (const float*)d_in[1];   // (B, N, 2) f32

    // Workspace carve-up (~68.2 MB total)
    char* ws = (char*)d_ws;
    _Float16* fnh = (_Float16*)ws; ws += (size_t)BATCH * NPTS * DIM * sizeof(_Float16);
    float* S      = (float*)ws;    ws += (size_t)BATCH * NPTS * NPTS * sizeof(float);
    float* cent   = (float*)ws;    ws += (size_t)BATCH * KCL * DIM * sizeof(float);
    int* ids      = (int*)ws;      ws += (size_t)BATCH * NPTS * sizeof(int);
    float* conn   = (float*)ws;    ws += (size_t)BATCH * NPTS * sizeof(float);
    int* order    = (int*)ws;      ws += (size_t)BATCH * NPTS * sizeof(int);

    float* out      = (float*)d_out;                       // (B, N, D)
    float* out_idx  = out + (size_t)BATCH * NPTS * DIM;    // (B, N)

    // 1) normalized f16 features for WMMA Gram
    normalize_f16_kernel<<<BATCH * NPTS, DIM, 0, stream>>>(feat, fnh);

    // 2-4) k-means (deterministic init; JAX PRNG not reproducible on-device)
    kmeans_init_kernel<<<BATCH * KCL, DIM, 0, stream>>>(feat, cent);
    for (int it = 0; it < KM_ITERS; ++it) {
        kmeans_assign_kernel<<<(BATCH * NPTS + 255) / 256, 256, 0, stream>>>(feat, cent, ids);
        kmeans_update_kernel<<<BATCH * KCL, DIM, 0, stream>>>(feat, ids, cent);
    }
    kmeans_assign_kernel<<<(BATCH * NPTS + 255) / 256, 256, 0, stream>>>(feat, cent, ids);

    // 5) similarity matrix: 65536 16x16 tiles, 8 waves/block
    gram_wmma_kernel<<<(BATCH * 128 * 128) / 8, 256, 0, stream>>>(fnh, coords, S);

    // 6) cluster connectivity
    conn_kernel<<<BATCH * NPTS, 256, 0, stream>>>(S, ids, conn);

    // 7) sequential ordering, one WG per batch
    order_kernel<<<BATCH, 1024, 0, stream>>>(S, conn, ids, order);

    // 8) gather output
    gather_kernel<<<BATCH * NPTS, 256, 0, stream>>>(feat, order, out, out_idx);
}
